// GCNClassifier_58909771432362
// MI455X (gfx1250) — matrix operand order repaired and checked by reading, not compiled, not run on hardware
//
#include <hip/hip_runtime.h>
#include <math.h>

// ---------------------------------------------------------------------------
// GCNClassifier for MI455X (gfx1250, wave32).
//   conv1 GEMM (100000x768 @ 768x128) -> bf16 WMMA (v_wmma_f32_16x16x32_bf16)
//   edge scatter-add -> global_atomic_add_f32 (L2-resident, 192MB L2)
//   conv2 / pool / log_softmax -> trivial VALU kernels
// Workspace usage ~105 MB: deg/dinv[N], hx[N*128], agg1[N*128], h2[N*2],
// agg2[N*2], pool[G*2], cnt[G].
// ---------------------------------------------------------------------------

typedef __attribute__((ext_vector_type(16))) __bf16 v16bf;
typedef __attribute__((ext_vector_type(8)))  float  v8f;

#define KDIM 768
#define HDIM 128
#define ODIM 2
#define BM   128
#define BK   32
#define LDK  (BK + 8)   // pad rows to 80B: keeps 16B alignment, staggers banks

// ---------------- small helper kernels ----------------

__global__ __launch_bounds__(256) void k_init_deg(float* __restrict__ deg, int n) {
  int i = blockIdx.x * 256 + threadIdx.x;
  if (i < n) deg[i] = 1.0f;                 // self-loop contributes 1
}

__global__ __launch_bounds__(256) void k_count_deg(const int* __restrict__ col,
                                                   float* __restrict__ deg, int e) {
  int i = blockIdx.x * 256 + threadIdx.x;
  if (i < e) atomicAdd(&deg[col[i]], 1.0f);
}

__global__ __launch_bounds__(256) void k_rsqrt(float* __restrict__ deg, int n) {
  int i = blockIdx.x * 256 + threadIdx.x;
  if (i < n) deg[i] = rsqrtf(deg[i]);       // deg -> dinv in place
}

// agg1[node][f] = b1[f]   (bias folded into aggregation init)
__global__ __launch_bounds__(256) void k_init_agg1(const float* __restrict__ b1,
                                                   float* __restrict__ agg1, int n) {
  int t = blockIdx.x * 256 + threadIdx.x;   // one float4 per thread
  if (t < n * 32) {
    float4 b = ((const float4*)b1)[t & 31];
    ((float4*)agg1)[t] = b;
  }
}

// ---------------- conv1 GEMM: hx = x @ W1 (bf16 WMMA, f32 accumulate) -------
// block tile 128x128 (full output width), 8 waves -> each wave 32x64,
// K-loop of 24 steps of BK=32 through LDS with fp32->bf16 conversion.

__global__ __launch_bounds__(256) void k_gemm1_wmma(const float* __restrict__ X,
                                                    const float* __restrict__ W,
                                                    float* __restrict__ Hx, int Nn) {
  __shared__ __bf16 lA[BM][LDK];            // A tile   [m][k]   (row-major)
  __shared__ __bf16 lB[HDIM][LDK];          // B tile   [n][k]   (transposed)

  const int t    = threadIdx.x;
  const int lane = t & 31;
  const int wave = t >> 5;                  // 0..7
  const int mb   = (wave >> 1) * 32;        // wave row base   (0,32,64,96)
  const int nb   = (wave & 1) * 64;         // wave col base   (0,64)
  const int rowBase = blockIdx.x * BM;

  const int half = lane >> 4;               // ISA lane-half
  const int l15  = lane & 15;

  v8f acc[2][4];
  #pragma unroll
  for (int i = 0; i < 2; ++i)
    #pragma unroll
    for (int j = 0; j < 4; ++j)
      #pragma unroll
      for (int v = 0; v < 8; ++v) acc[i][j][v] = 0.0f;

  // A-tile loader mapping: thread t covers row t>>1, 16 contiguous k's
  const int arow = t >> 1;
  const int aks  = (t & 1) * 16;
  // B-tile loader mapping: thread t covers column n = t&127, k rows t>>7 + kk (kk even)
  const int bn   = t & 127;
  const int bk0  = t >> 7;

  for (int k0 = 0; k0 < KDIM; k0 += BK) {
    // ---- stage A tile (convert fp32 -> bf16) ----
    {
      int grow = rowBase + arow;
      if (grow < Nn) {
        const float* src = X + (size_t)grow * KDIM + k0 + aks;
        #pragma unroll
        for (int i = 0; i < 16; i += 4) {
          float4 v = *(const float4*)(src + i);
          lA[arow][aks + i + 0] = (__bf16)v.x;
          lA[arow][aks + i + 1] = (__bf16)v.y;
          lA[arow][aks + i + 2] = (__bf16)v.z;
          lA[arow][aks + i + 3] = (__bf16)v.w;
        }
      } else {
        #pragma unroll
        for (int i = 0; i < 16; ++i) lA[arow][aks + i] = (__bf16)0.0f;
      }
    }
    // ---- stage B tile transposed: lB[n][k] = W1[k0+k][n] (coalesced reads) --
    #pragma unroll
    for (int kk = 0; kk < BK; kk += 2) {
      int k = kk + bk0;
      lB[bn][k] = (__bf16)W[(size_t)(k0 + k) * HDIM + bn];
    }
    __syncthreads();

    // ---- build fragments per ISA 16-bit layouts ----
    // A 16x32: lane m = l15, VGPR g<4 -> K=2g+{0,1}+8*half ; g>=4 -> +16
    //          => elems 0..7  : K = 8*half + e
    //             elems 8..15 : K = 16 + 8*half + (e-8)      (two 16B runs)
    v16bf af[2];
    #pragma unroll
    for (int i = 0; i < 2; ++i) {
      const __bf16* p = &lA[mb + 16 * i + l15][0];
      #pragma unroll
      for (int e = 0; e < 8; ++e) af[i][e]     = p[8 * half + e];
      #pragma unroll
      for (int e = 0; e < 8; ++e) af[i][8 + e] = p[16 + 8 * half + e];
    }
    // B 32x16: lane n = l15, same K order as A
    //          => elems 0..7 : K = 8*half + e ; elems 8..15 : K = 16 + 8*half + (e-8)
    v16bf bfr[4];
    #pragma unroll
    for (int j = 0; j < 4; ++j) {
      const __bf16* p = &lB[nb + 16 * j + l15][0];
      #pragma unroll
      for (int e = 0; e < 8; ++e) bfr[j][e] = p[8 * half + e];
      #pragma unroll
      for (int e = 0; e < 8; ++e) bfr[j][8 + e] = p[16 + 8 * half + e];
    }

    #pragma unroll
    for (int i = 0; i < 2; ++i)
      #pragma unroll
      for (int j = 0; j < 4; ++j)
        acc[i][j] = __builtin_amdgcn_wmma_f32_16x16x32_bf16(
            false, af[i], false, bfr[j], (short)0, acc[i][j], false, false);

    __syncthreads();
  }

  // ---- epilogue: C VGPR v -> M = v + 8*half, N = l15 ----
  #pragma unroll
  for (int i = 0; i < 2; ++i)
    #pragma unroll
    for (int j = 0; j < 4; ++j)
      #pragma unroll
      for (int v = 0; v < 8; ++v) {
        int grow = rowBase + mb + 16 * i + v + 8 * half;
        int gcol = nb + 16 * j + l15;
        if (grow < Nn) Hx[(size_t)grow * HDIM + gcol] = acc[i][j][v];
      }
}

// ---------------- conv1 scatter: agg1[col] += hx[row]*norm ------------------
// one wave per edge, 4 features per lane (float4 gather + 4 f32 atomics)

__global__ __launch_bounds__(256) void k_scatter1(const int* __restrict__ row,
                                                  const int* __restrict__ col,
                                                  const float* __restrict__ dinv,
                                                  const float* __restrict__ hx,
                                                  float* __restrict__ agg1,
                                                  int e, int n) {
  unsigned long long tid = (unsigned long long)blockIdx.x * 256 + threadIdx.x;
  unsigned int edge = (unsigned int)(tid >> 5);
  int lane = (int)(tid & 31);
  unsigned int en = (unsigned int)(e + n);
  if (edge >= en) return;
  int r, c;
  if (edge < (unsigned int)e) { r = row[edge]; c = col[edge]; }
  else                        { r = c = (int)(edge - e); }      // self loop
  float norm = dinv[r] * dinv[c];
  float4 v = *(const float4*)(hx + (size_t)r * HDIM + lane * 4);
  float* dst = agg1 + (size_t)c * HDIM + lane * 4;
  atomicAdd(dst + 0, v.x * norm);
  atomicAdd(dst + 1, v.y * norm);
  atomicAdd(dst + 2, v.z * norm);
  atomicAdd(dst + 3, v.w * norm);
}

// ---------------- ReLU + conv2 GEMM (128 -> 2) + agg2 bias init -------------

__global__ __launch_bounds__(256) void k_relu_gemm2(const float* __restrict__ agg1,
                                                    const float* __restrict__ W2,
                                                    const float* __restrict__ b2,
                                                    float* __restrict__ h2,
                                                    float* __restrict__ agg2, int n) {
  int i = blockIdx.x * 256 + threadIdx.x;
  if (i >= n) return;
  const float4* a = (const float4*)(agg1 + (size_t)i * HDIM);
  float acc0 = 0.0f, acc1 = 0.0f;
  #pragma unroll
  for (int q = 0; q < HDIM / 4; ++q) {
    float4 v = a[q];
    float r0 = fmaxf(v.x, 0.0f), r1 = fmaxf(v.y, 0.0f);
    float r2 = fmaxf(v.z, 0.0f), r3 = fmaxf(v.w, 0.0f);
    int f = q * 4;
    acc0 += r0 * W2[(f + 0) * 2] + r1 * W2[(f + 1) * 2] +
            r2 * W2[(f + 2) * 2] + r3 * W2[(f + 3) * 2];
    acc1 += r0 * W2[(f + 0) * 2 + 1] + r1 * W2[(f + 1) * 2 + 1] +
            r2 * W2[(f + 2) * 2 + 1] + r3 * W2[(f + 3) * 2 + 1];
  }
  h2[2 * i]     = acc0;
  h2[2 * i + 1] = acc1;
  agg2[2 * i]     = b2[0];
  agg2[2 * i + 1] = b2[1];
}

// ---------------- conv2 scatter: agg2[col] += h2[row]*norm ------------------

__global__ __launch_bounds__(256) void k_scatter2(const int* __restrict__ row,
                                                  const int* __restrict__ col,
                                                  const float* __restrict__ dinv,
                                                  const float* __restrict__ h2,
                                                  float* __restrict__ agg2,
                                                  int e, int n) {
  int edge = blockIdx.x * 256 + threadIdx.x;
  if (edge >= e + n) return;
  int r, c;
  if (edge < e) { r = row[edge]; c = col[edge]; }
  else          { r = c = edge - e; }
  float norm = dinv[r] * dinv[c];
  atomicAdd(agg2 + 2 * c,     h2[2 * r]     * norm);
  atomicAdd(agg2 + 2 * c + 1, h2[2 * r + 1] * norm);
}

// ---------------- mean pool + log_softmax -----------------------------------

__global__ __launch_bounds__(256) void k_zero_pool(float* __restrict__ pool,
                                                   float* __restrict__ cnt, int g) {
  for (int t = threadIdx.x; t < 2 * g; t += 256) pool[t] = 0.0f;
  for (int t = threadIdx.x; t < g; t += 256) cnt[t] = 0.0f;
}

__global__ __launch_bounds__(256) void k_pool(const int* __restrict__ batch,
                                              const float* __restrict__ agg2,
                                              float* __restrict__ pool,
                                              float* __restrict__ cnt, int n) {
  int i = blockIdx.x * 256 + threadIdx.x;
  if (i >= n) return;
  int g = batch[i];
  atomicAdd(pool + 2 * g,     agg2[2 * i]);
  atomicAdd(pool + 2 * g + 1, agg2[2 * i + 1]);
  atomicAdd(cnt + g, 1.0f);
}

__global__ __launch_bounds__(256) void k_logsoftmax(const float* __restrict__ pool,
                                                    const float* __restrict__ cnt,
                                                    float* __restrict__ out, int g) {
  int i = blockIdx.x * 256 + threadIdx.x;
  if (i >= g) return;
  float c  = fmaxf(cnt[i], 1.0f);
  float s0 = pool[2 * i] / c;
  float s1 = pool[2 * i + 1] / c;
  float m  = fmaxf(s0, s1);
  float lse = m + logf(expf(s0 - m) + expf(s1 - m));
  out[2 * i]     = s0 - lse;
  out[2 * i + 1] = s1 - lse;
}

// ---------------------------------------------------------------------------

extern "C" void kernel_launch(void* const* d_in, const int* in_sizes, int n_in,
                              void* d_out, int out_size, void* d_ws, size_t ws_size,
                              hipStream_t stream) {
  const float* x     = (const float*)d_in[0];
  const int*   ei    = (const int*)d_in[1];     // [2, E] flat
  const int*   batch = (const int*)d_in[2];
  const float* W1    = (const float*)d_in[3];
  const float* b1    = (const float*)d_in[4];
  const float* W2    = (const float*)d_in[5];
  const float* b2    = (const float*)d_in[6];
  float* out = (float*)d_out;

  const int Nn = in_sizes[2];                   // 100000 nodes
  const int Ee = in_sizes[1] / 2;               // 3.2M edges
  const int Gg = out_size / ODIM;               // 128 graphs
  const int EN = Ee + Nn;                       // edges incl. self loops

  const int* row = ei;
  const int* col = ei + Ee;

  // ---- carve workspace (~105 MB) ----
  float* ws = (float*)d_ws;
  size_t off = 0;
  auto alloc = [&](size_t elems) { float* p = ws + off; off += (elems + 3) & ~(size_t)3; return p; };
  float* deg  = alloc((size_t)Nn);              // then dinv (in place)
  float* hx   = alloc((size_t)Nn * HDIM);
  float* agg1 = alloc((size_t)Nn * HDIM);
  float* h2   = alloc((size_t)Nn * ODIM);
  float* agg2 = alloc((size_t)Nn * ODIM);
  float* pool = alloc((size_t)Gg * ODIM);
  float* cnt  = alloc((size_t)Gg);
  (void)ws_size; (void)n_in;

  const int B = 256;
  // degrees / norms
  k_init_deg <<<(Nn + B - 1) / B, B, 0, stream>>>(deg, Nn);
  k_count_deg<<<(Ee + B - 1) / B, B, 0, stream>>>(col, deg, Ee);
  k_rsqrt    <<<(Nn + B - 1) / B, B, 0, stream>>>(deg, Nn);
  // conv1
  k_init_agg1<<<((size_t)Nn * 32 + B - 1) / B, B, 0, stream>>>(b1, agg1, Nn);
  k_gemm1_wmma<<<(Nn + BM - 1) / BM, B, 0, stream>>>(x, W1, hx, Nn);
  k_scatter1 <<<(unsigned)(((unsigned long long)EN * 32 + B - 1) / B), B, 0, stream>>>(
      row, col, deg, hx, agg1, Ee, Nn);
  // relu + conv2
  k_relu_gemm2<<<(Nn + B - 1) / B, B, 0, stream>>>(agg1, W2, b2, h2, agg2, Nn);
  k_scatter2  <<<(EN + B - 1) / B, B, 0, stream>>>(row, col, deg, h2, agg2, Ee, Nn);
  // mean pool + log_softmax
  k_zero_pool <<<1, B, 0, stream>>>(pool, cnt, Gg);
  k_pool      <<<(Nn + B - 1) / B, B, 0, stream>>>(batch, agg2, pool, cnt, Nn);
  k_logsoftmax<<<(Gg + B - 1) / B, B, 0, stream>>>(pool, cnt, out, Gg);
}
